// LSTMClassifier_29583734735528
// MI455X (gfx1250) — matrix operand order repaired and checked by reading, not compiled, not run on hardware
//
#include <hip/hip_runtime.h>
#include <hip/hip_bf16.h>
#include <math.h>

// ---------------------------------------------------------------------------
// Types for CDNA5 WMMA (gfx1250, wave32)
// ---------------------------------------------------------------------------
typedef __attribute__((ext_vector_type(16))) __bf16 v16bf;
typedef __attribute__((ext_vector_type(8)))  __bf16 v8bf;
typedef __attribute__((ext_vector_type(8)))  float  v8f;

#define LSTM_B 64
#define LSTM_T 512
#define LSTM_H 512
#define GDIM   2048   // 4*H

union FragBF { v16bf v; v8bf h[2]; };
union BFBits { unsigned short u; __bf16 b; };

__device__ __forceinline__ __bf16 f2bf(float f) {
  unsigned u = __float_as_uint(f);
  unsigned r = (u + 0x7FFFu + ((u >> 16) & 1u)) >> 16;  // RNE
  BFBits bb; bb.u = (unsigned short)r;
  return bb.b;
}
__device__ __forceinline__ float bf2f(__bf16 b) {
  BFBits bb; bb.b = b;
  return __uint_as_float(((unsigned)bb.u) << 16);
}
__device__ __forceinline__ float sigmoidf_(float x) {
  return 1.0f / (1.0f + __expf(-x));
}
__device__ __forceinline__ float tanhf_(float x) {
  float ax = fabsf(x);
  float e  = __expf(-2.0f * ax);
  float t  = (1.0f - e) / (1.0f + e);
  return copysignf(t, x);
}

// A fragment (16x32 bf16, MxK): lane m (0-15) holds row m.
// VGPR0-3: K = koff..koff+7, VGPR4-7: K = koff+16..koff+23, koff = 8*(lane>=16).
__device__ __forceinline__ v16bf load_A(const __bf16* rowptr, int k, int lane) {
  int koff = (lane >> 4) << 3;
  const __bf16* p = rowptr + k + koff;
  FragBF f;
  f.h[0] = *(const v8bf*)(p);
  f.h[1] = *(const v8bf*)(p + 16);
  return f.v;
}

// B fragment (32x16 bf16, KxN) from row-major weight W (rows = output features,
// ld = K stride): lane holds column (lane&15), K = kb..kb+7 and kb+16..kb+23,
// kb = 8*(lane>=16).
__device__ __forceinline__ v16bf load_B(const __bf16* w_tile, size_t ld, int k, int lane) {
  int n  = lane & 15;
  int kb = (lane >> 4) << 3;
  const __bf16* p = w_tile + (size_t)n * ld + k + kb;
  FragBF f;
  f.h[0] = *(const v8bf*)(p);
  f.h[1] = *(const v8bf*)(p + 16);
  return f.v;
}

#define WMMA_BF16(ACC, A, Bm) \
  (ACC) = __builtin_amdgcn_wmma_f32_16x16x32_bf16(false, (A), false, (Bm), (short)0, (ACC), false, false)

// ---------------------------------------------------------------------------
// One LSTM timestep (both directions via blockIdx.y). Fuses x-projection and
// recurrent matmul: g = bias + x_t @ W_ih^T + h @ W_hh^T, then gate update.
// Each wave owns a 16(batch) x 16(h) tile across all 4 gates.
// ---------------------------------------------------------------------------
template <int DIN>
__global__ void __launch_bounds__(256)
lstm_step_kernel(const __bf16* __restrict__ xin,     // input sequence (bf16)
                 size_t x_row_stride,                 // elems between batch rows
                 size_t x_t_stride,                   // elems between timesteps
                 const __bf16* __restrict__ w_ih,     // (2, GDIM, DIN)
                 const __bf16* __restrict__ w_hh,     // (2, GDIM, H)
                 const float*  __restrict__ bsum,     // (2, GDIM)  b_ih + b_hh
                 const __bf16* __restrict__ h_in,     // (2, B, H)
                 __bf16*       __restrict__ h_out,    // (2, B, H)
                 float*        __restrict__ c_state,  // (2, B, H)
                 __bf16*       __restrict__ outseq,   // (B, T, 2H)
                 int t)
{
  const int B = LSTM_B, T = LSTM_T, H = LSTM_H;
  const int d    = blockIdx.y;                               // direction
  const int lane = threadIdx.x & 31;
  const int wid  = blockIdx.x * (blockDim.x >> 5) + (threadIdx.x >> 5); // 0..127
  const int b0   = (wid >> 5) << 4;                          // batch tile base
  const int h0   = (wid & 31) << 4;                          // hidden tile base
  const int t_d  = d ? (T - 1 - t) : t;

  const int nl = lane & 15;   // column within tile (for bias / D layout)
  const int rA = lane & 15;   // A row within tile for this lane

  // accumulators, one per gate, initialized with bias broadcast along M
  const float* bs = bsum + (size_t)d * GDIM + h0 + nl;
  v8f acc0, acc1, acc2, acc3;
  {
    float v0 = bs[0 * H], v1 = bs[1 * H], v2 = bs[2 * H], v3 = bs[3 * H];
    #pragma unroll
    for (int r = 0; r < 8; ++r) { acc0[r] = v0; acc1[r] = v1; acc2[r] = v2; acc3[r] = v3; }
  }

  // ---- input projection part: K over DIN ----
  {
    const __bf16* xrow = xin + (size_t)(b0 + rA) * x_row_stride + (size_t)t_d * x_t_stride;
    const __bf16* wi = w_ih + (size_t)d * GDIM * DIN + (size_t)h0 * DIN;
    #pragma unroll 4
    for (int k = 0; k < DIN; k += 32) {
      v16bf a  = load_A(xrow, k, lane);
      v16bf bw0 = load_B(wi + (size_t)(0 * H) * DIN, DIN, k, lane);
      v16bf bw1 = load_B(wi + (size_t)(1 * H) * DIN, DIN, k, lane);
      v16bf bw2 = load_B(wi + (size_t)(2 * H) * DIN, DIN, k, lane);
      v16bf bw3 = load_B(wi + (size_t)(3 * H) * DIN, DIN, k, lane);
      WMMA_BF16(acc0, a, bw0);
      WMMA_BF16(acc1, a, bw1);
      WMMA_BF16(acc2, a, bw2);
      WMMA_BF16(acc3, a, bw3);
    }
  }

  // ---- recurrent part: K over H ----
  {
    const __bf16* hrow = h_in + ((size_t)d * B + b0 + rA) * H;
    const __bf16* wh = w_hh + (size_t)d * GDIM * H + (size_t)h0 * H;
    #pragma unroll 4
    for (int k = 0; k < H; k += 32) {
      v16bf a  = load_A(hrow, k, lane);
      v16bf bw0 = load_B(wh + (size_t)(0 * H) * H, H, k, lane);
      v16bf bw1 = load_B(wh + (size_t)(1 * H) * H, H, k, lane);
      v16bf bw2 = load_B(wh + (size_t)(2 * H) * H, H, k, lane);
      v16bf bw3 = load_B(wh + (size_t)(3 * H) * H, H, k, lane);
      WMMA_BF16(acc0, a, bw0);
      WMMA_BF16(acc1, a, bw1);
      WMMA_BF16(acc2, a, bw2);
      WMMA_BF16(acc3, a, bw3);
    }
  }

  // ---- gates + state update (in-lane, per ISA C/D layout) ----
  const int mb = (lane >> 4) << 3;   // row base: 0 or 8
  #pragma unroll
  for (int r = 0; r < 8; ++r) {
    const int b  = b0 + mb + r;
    const int hh = h0 + nl;
    const size_t sidx = ((size_t)d * B + b) * H + hh;
    float iv = sigmoidf_(acc0[r]);
    float fv = sigmoidf_(acc1[r]);
    float gv = tanhf_(acc2[r]);
    float ov = sigmoidf_(acc3[r]);
    float c  = fv * c_state[sidx] + iv * gv;
    c_state[sidx] = c;
    float hv = ov * tanhf_(c);
    __bf16 hb = f2bf(hv);
    h_out[sidx] = hb;
    outseq[((size_t)b * T + t_d) * (2 * H) + (size_t)d * H + hh] = hb;
  }
}

// ---------------------------------------------------------------------------
// Helpers: f32 -> bf16 conversion, bias sum
// ---------------------------------------------------------------------------
__global__ void cvt_bf16_kernel(const float* __restrict__ in, __bf16* __restrict__ out, size_t n) {
  size_t i = (size_t)blockIdx.x * blockDim.x + threadIdx.x;
  if (i < n) out[i] = f2bf(in[i]);
}

__global__ void bias_sum_kernel(const float* __restrict__ a, const float* __restrict__ b,
                                float* __restrict__ o, int n) {
  int i = blockIdx.x * blockDim.x + threadIdx.x;
  if (i < n) o[i] = a[i] + b[i];
}

// ---------------------------------------------------------------------------
// Head: layernorm over last timestep, 512x1024 matvec + ReLU, 1x512 matvec.
// One block (256 threads) per batch element.
// ---------------------------------------------------------------------------
__global__ void __launch_bounds__(256)
head_kernel(const __bf16* __restrict__ out1, const float* __restrict__ ln_g,
            const float* __restrict__ ln_b, const float* __restrict__ w1,
            const float* __restrict__ b1, const float* __restrict__ w2,
            const float* __restrict__ b2, float* __restrict__ out) {
  __shared__ float vbuf[1024];
  __shared__ float red[256];
  const int T = LSTM_T;
  const int b = blockIdx.x;
  const int tid = threadIdx.x;

  float s = 0.f, ss = 0.f;
  for (int j = tid; j < 1024; j += 256) {
    float f = bf2f(out1[((size_t)b * T + (T - 1)) * 1024 + j]);
    vbuf[j] = f; s += f; ss += f * f;
  }
  red[tid] = s; __syncthreads();
  for (int o = 128; o > 0; o >>= 1) { if (tid < o) red[tid] += red[tid + o]; __syncthreads(); }
  float mean = red[0] * (1.f / 1024.f);
  __syncthreads();
  red[tid] = ss; __syncthreads();
  for (int o = 128; o > 0; o >>= 1) { if (tid < o) red[tid] += red[tid + o]; __syncthreads(); }
  float var  = red[0] * (1.f / 1024.f) - mean * mean;
  float rstd = rsqrtf(var + 1e-5f);
  __syncthreads();
  for (int j = tid; j < 1024; j += 256)
    vbuf[j] = (vbuf[j] - mean) * rstd * ln_g[j] + ln_b[j];
  __syncthreads();

  float acc = 0.f;
  for (int i = tid; i < 512; i += 256) {
    const float* wr = w1 + (size_t)i * 1024;
    float a = b1[i];
    #pragma unroll 4
    for (int j = 0; j < 1024; ++j) a += vbuf[j] * wr[j];
    a = fmaxf(a, 0.f);
    acc += a * w2[i];
  }
  red[tid] = acc; __syncthreads();
  for (int o = 128; o > 0; o >>= 1) { if (tid < o) red[tid] += red[tid + o]; __syncthreads(); }
  if (tid == 0) out[b] = red[0] + b2[0];
}

// ---------------------------------------------------------------------------
// Launch
// ---------------------------------------------------------------------------
extern "C" void kernel_launch(void* const* d_in, const int* in_sizes, int n_in,
                              void* d_out, int out_size, void* d_ws, size_t ws_size,
                              hipStream_t stream) {
  const float* x    = (const float*)d_in[0];
  const float* wih0 = (const float*)d_in[1];
  const float* whh0 = (const float*)d_in[2];
  const float* bih0 = (const float*)d_in[3];
  const float* bhh0 = (const float*)d_in[4];
  const float* wih1 = (const float*)d_in[5];
  const float* whh1 = (const float*)d_in[6];
  const float* bih1 = (const float*)d_in[7];
  const float* bhh1 = (const float*)d_in[8];
  const float* ln_g = (const float*)d_in[9];
  const float* ln_b = (const float*)d_in[10];
  const float* w1   = (const float*)d_in[11];
  const float* b1   = (const float*)d_in[12];
  const float* w2   = (const float*)d_in[13];
  const float* b2   = (const float*)d_in[14];
  float* out = (float*)d_out;

  const int B = LSTM_B, T = LSTM_T, H = LSTM_H, G = GDIM;
  const int D0 = 256, D1 = 1024;

  // workspace carve-up (~170 MB total), 256B aligned
  char* p = (char*)d_ws;
  auto alloc = [&](size_t bytes) {
    char* r = p; p += (bytes + 255) & ~(size_t)255; return r;
  };
  __bf16* x_bf    = (__bf16*)alloc((size_t)B * T * D0 * 2);
  __bf16* wih0_bf = (__bf16*)alloc((size_t)2 * G * D0 * 2);
  __bf16* whh0_bf = (__bf16*)alloc((size_t)2 * G * H  * 2);
  __bf16* wih1_bf = (__bf16*)alloc((size_t)2 * G * D1 * 2);
  __bf16* whh1_bf = (__bf16*)alloc((size_t)2 * G * H  * 2);
  float*  bsum0   = (float*) alloc((size_t)2 * G * 4);
  float*  bsum1   = (float*) alloc((size_t)2 * G * 4);
  __bf16* out0    = (__bf16*)alloc((size_t)B * T * 2 * H * 2);
  __bf16* out1    = (__bf16*)alloc((size_t)B * T * 2 * H * 2);
  __bf16* hbuf    = (__bf16*)alloc((size_t)2 * 2 * B * H * 2);  // double-buffered
  float*  cbuf    = (float*) alloc((size_t)2 * B * H * 4);

  auto cvt = [&](const float* src, __bf16* dst, size_t n) {
    int blocks = (int)((n + 255) / 256);
    cvt_bf16_kernel<<<dim3(blocks), dim3(256), 0, stream>>>(src, dst, n);
  };
  cvt(x,    x_bf,    (size_t)B * T * D0);
  cvt(wih0, wih0_bf, (size_t)2 * G * D0);
  cvt(whh0, whh0_bf, (size_t)2 * G * H);
  cvt(wih1, wih1_bf, (size_t)2 * G * D1);
  cvt(whh1, whh1_bf, (size_t)2 * G * H);
  bias_sum_kernel<<<dim3((2 * G + 255) / 256), dim3(256), 0, stream>>>(bih0, bhh0, bsum0, 2 * G);
  bias_sum_kernel<<<dim3((2 * G + 255) / 256), dim3(256), 0, stream>>>(bih1, bhh1, bsum1, 2 * G);

  const size_t hstride = (size_t)2 * B * H;  // elems per h parity buffer

  // ---- layer 0 ----
  hipMemsetAsync(hbuf, 0, (size_t)2 * hstride * 2, stream);
  hipMemsetAsync(cbuf, 0, (size_t)2 * B * H * 4, stream);
  for (int t = 0; t < T; ++t) {
    const __bf16* h_in = hbuf + (size_t)(t & 1) * hstride;
    __bf16* h_out      = hbuf + (size_t)((t + 1) & 1) * hstride;
    lstm_step_kernel<256><<<dim3(16, 2), dim3(256), 0, stream>>>(
        x_bf, (size_t)T * D0, (size_t)D0,
        wih0_bf, whh0_bf, bsum0, h_in, h_out, cbuf, out0, t);
  }

  // ---- layer 1 ----
  hipMemsetAsync(hbuf, 0, (size_t)2 * hstride * 2, stream);
  hipMemsetAsync(cbuf, 0, (size_t)2 * B * H * 4, stream);
  for (int t = 0; t < T; ++t) {
    const __bf16* h_in = hbuf + (size_t)(t & 1) * hstride;
    __bf16* h_out      = hbuf + (size_t)((t + 1) & 1) * hstride;
    lstm_step_kernel<1024><<<dim3(16, 2), dim3(256), 0, stream>>>(
        out0, (size_t)T * D1, (size_t)D1,
        wih1_bf, whh1_bf, bsum1, h_in, h_out, cbuf, out1, t);
  }

  // ---- head ----
  head_kernel<<<dim3(B), dim3(256), 0, stream>>>(out1, ln_g, ln_b, w1, b1, w2, b2, out);
}
